// SinkhornSelfAttention_48533130445203
// MI455X (gfx1250) — matrix operand order repaired and checked by reading, not compiled, not run on hardware
//
#include <hip/hip_runtime.h>

// ---------- types ----------
typedef __attribute__((ext_vector_type(16))) __bf16 bf16x16;
typedef __attribute__((ext_vector_type(8)))  __bf16 bf16x8;
typedef __attribute__((ext_vector_type(8)))  float  f32x8;

// problem constants
#define N_B      4
#define T_SEQ    4096
#define D_MODEL  1024
#define N_H      8
#define D_HEAD   128
#define N_BUCKET 64
#define BSZ      64
#define BHDIM    32          // N_B * N_H
#define ROWS     16384       // N_B * T_SEQ

__device__ __forceinline__ f32x8 zero8() {
  f32x8 z;
#pragma unroll
  for (int e = 0; e < 8; ++e) z[e] = 0.0f;
  return z;
}

__device__ __forceinline__ f32x8 wmma_bf16(bf16x16 a, bf16x16 b, f32x8 c) {
  return __builtin_amdgcn_wmma_f32_16x16x32_bf16(false, a, false, b, (short)0, c,
                                                 false, false);
}

// CDNA5 async copy: 16 bytes global -> LDS, tracked by ASYNCcnt.
__device__ __forceinline__ void async_copy16(unsigned lds_off, const void* gaddr) {
  asm volatile("global_load_async_to_lds_b128 %0, %1, off"
               :: "v"(lds_off), "v"(gaddr)
               : "memory");
}

// A fragment: 16(M) x 32(K) bf16, src row-major [m][k]; stride in elements.
// Lane L: m=L&15, half=L>>4; VGPR v holds K=half*16 + 2v,2v+1
__device__ __forceinline__ bf16x16 load_a_frag(const __bf16* base, int stride, int lane) {
  int m = lane & 15, half = lane >> 4;
  const __bf16* p = base + m * stride + half * 16;
  union { bf16x16 v; bf16x8 h[2]; } u;
  u.h[0] = *(const bf16x8*)(p);
  u.h[1] = *(const bf16x8*)(p + 8);
  return u.v;
}

// B fragment: 32(K) x 16(N) bf16, src stored transposed [n][k] row-major.
// Lane L: n=L&15, half=L>>4; VGPR v holds K=half*16 + 2v,2v+1
__device__ __forceinline__ bf16x16 load_b_frag(const __bf16* base, int stride, int lane) {
  int n = lane & 15, half = lane >> 4;
  return *(const bf16x16*)(base + n * stride + half * 16);
}

// ---------- 1: fp32 -> bf16 cast ----------
__global__ void k_cast_bf16(const float* __restrict__ in, __bf16* __restrict__ out, int n) {
  int i = blockIdx.x * blockDim.x + threadIdx.x;
  if (i < n) out[i] = (__bf16)in[i];
}

// ---------- 2: QKV GEMM: [16384x1024] x [3072x1024]^T, scatter to bucketed q/k/v ----
// Double-buffered LDS fed by GLOBAL_LOAD_ASYNC_TO_LDS_B128 (ASYNCcnt pipeline).
__global__ __launch_bounds__(256) void k_gemm_qkv(
    const __bf16* __restrict__ A, const __bf16* __restrict__ W,
    __bf16* __restrict__ qb, __bf16* __restrict__ kb, __bf16* __restrict__ vb) {
  const int K = D_MODEL;
  __shared__ __bf16 sA[2][128 * 32];
  __shared__ __bf16 sB[2][128 * 32];
  int tid = threadIdx.x, lane = tid & 31, wave = tid >> 5;
  int wm = wave & 1, wn = wave >> 1;            // wave tile: 64 rows x 32 cols
  int row0 = blockIdx.y * 128, col0 = blockIdx.x * 128;
  int lr = tid >> 1, lh = tid & 1;

  f32x8 acc[4][2];
#pragma unroll
  for (int i = 0; i < 4; ++i)
#pragma unroll
    for (int j = 0; j < 2; ++j) acc[i][j] = zero8();

  const __bf16* gA = &A[(size_t)(row0 + lr) * K + lh * 16];
  const __bf16* gW = &W[(size_t)(col0 + lr) * K + lh * 16];

  // issue 4 x 16B async copies (A lo/hi, B lo/hi) for k-step k0 into buffer `buf`
  auto issue = [&](int buf, int k0) {
    unsigned la = (unsigned)(size_t)&sA[buf][lr * 32 + lh * 16];
    unsigned lb = (unsigned)(size_t)&sB[buf][lr * 32 + lh * 16];
    async_copy16(la,      gA + k0);
    async_copy16(la + 16, gA + k0 + 8);
    async_copy16(lb,      gW + k0);
    async_copy16(lb + 16, gW + k0 + 8);
  };

  const int NIT = K / 32;
  issue(0, 0);
  for (int it = 0; it < NIT; ++it) {
    int cur = it & 1;
    if (it + 1 < NIT) {
      issue(cur ^ 1, (it + 1) * 32);
      asm volatile("s_wait_asynccnt 0x4" ::: "memory");  // current buffer landed
    } else {
      asm volatile("s_wait_asynccnt 0x0" ::: "memory");
    }
    __syncthreads();
    bf16x16 bf[2];
#pragma unroll
    for (int j = 0; j < 2; ++j)
      bf[j] = load_b_frag(&sB[cur][(wn * 32 + j * 16) * 32], 32, lane);
#pragma unroll
    for (int i = 0; i < 4; ++i) {
      bf16x16 af = load_a_frag(&sA[cur][(wm * 64 + i * 16) * 32], 32, lane);
#pragma unroll
      for (int j = 0; j < 2; ++j) acc[i][j] = wmma_bf16(af, bf[j], acc[i][j]);
    }
    __syncthreads();  // all waves done reading buf before it is overwritten
  }

  int nloc = lane & 15, mhalf = (lane >> 4) * 8;
#pragma unroll
  for (int i = 0; i < 4; ++i) {
#pragma unroll
    for (int j = 0; j < 2; ++j) {
      int c = col0 + wn * 32 + j * 16 + nloc;
      int which = c >> 10, dp = c & 1023;
      int h = dp >> 7, dd = dp & 127;
      __bf16* dst = (which == 0) ? qb : (which == 1) ? kb : vb;
#pragma unroll
      for (int e = 0; e < 8; ++e) {
        int r = row0 + wm * 64 + i * 16 + mhalf + e;
        int bb = r >> 12, tt = r & 4095;
        dst[(size_t)((bb * N_H + h) * T_SEQ + tt) * D_HEAD + dd] = (__bf16)acc[i][j][e];
      }
    }
  }
}

// ---------- 3: bucket sums + sinkhorn (1 block per bh) ----------
__global__ __launch_bounds__(256) void k_sinkhorn(
    const __bf16* __restrict__ kb, const float* __restrict__ sort_w,
    const float* __restrict__ noise_u, float* __restrict__ Rout) {
  __shared__ float sBKR[64 * 128];
  __shared__ float sR[64 * 64];
  int bh = blockIdx.x, tid = threadIdx.x;
  int h = bh & (N_H - 1);
  // bucket key sums: b_k_r[u][d]
  for (int o = tid; o < 64 * 128; o += 256) {
    int u = o >> 7, d = o & 127;
    const __bf16* p = kb + ((size_t)(bh * T_SEQ + u * BSZ)) * D_HEAD + d;
    float s = 0.0f;
    for (int tl = 0; tl < BSZ; ++tl) s += (float)p[tl * D_HEAD];
    sBKR[o] = s;
  }
  __syncthreads();
  // logits + gumbel
  for (int o = tid; o < 64 * 64; o += 256) {
    int u = o >> 6, v = o & 63;
    float dot = 0.0f;
    for (int d = 0; d < 128; ++d)
      dot += sBKR[u * 128 + d] * sort_w[(h * 128 + d) * 64 + v];
    dot = fmaxf(dot, 0.0f);
    float logit = logf(dot + 1e-6f);
    float nu = noise_u[(bh * 64 + u) * 64 + v];
    float g = -logf(-logf(nu + 1e-4f) + 1e-4f);
    sR[o] = (logit + g) * (1.0f / 0.75f);
  }
  __syncthreads();
  // 5 sinkhorn iterations in log space
  for (int it = 0; it < 5; ++it) {
    if (tid < 64) {  // rows (axis=2)
      float mx = -__builtin_inff();
      for (int v = 0; v < 64; ++v) mx = fmaxf(mx, sR[tid * 64 + v]);
      float s = 0.0f;
      for (int v = 0; v < 64; ++v) s += expf(sR[tid * 64 + v] - mx);
      float lse = mx + logf(s);
      for (int v = 0; v < 64; ++v) sR[tid * 64 + v] -= lse;
    }
    __syncthreads();
    if (tid < 64) {  // cols (axis=1)
      float mx = -__builtin_inff();
      for (int u = 0; u < 64; ++u) mx = fmaxf(mx, sR[u * 64 + tid]);
      float s = 0.0f;
      for (int u = 0; u < 64; ++u) s += expf(sR[u * 64 + tid] - mx);
      float lse = mx + logf(s);
      for (int u = 0; u < 64; ++u) sR[u * 64 + tid] -= lse;
    }
    __syncthreads();
  }
  // strictly-lower-triangular exp
  for (int o = tid; o < 64 * 64; o += 256) {
    int u = o >> 6, v = o & 63;
    Rout[bh * 4096 + o] = (u > v) ? expf(sR[o]) : 0.0f;
  }
}

// ---------- 4: soft permute: dst[bh][u][tl][dd] = sum_v R[u][v]*src[bh][v][tl][dd] ----
// grid: (64 tl-tiles, 32 bh). M=64(u), N=128(dd at fixed tl), K=64(v).
__global__ __launch_bounds__(256) void k_permute(
    const float* __restrict__ R, const __bf16* __restrict__ src,
    __bf16* __restrict__ dst) {
  __shared__ __bf16 sR[64 * 64];
  __shared__ __bf16 sBT[128 * 32];  // [n=dd][k=v]
  int bh = blockIdx.y, tl = blockIdx.x, tid = threadIdx.x;
  int lane = tid & 31, wave = tid >> 5;
  int wm = wave & 3, wn = wave >> 2;  // wave: 16 rows x 64 cols
  const float* Rp = R + bh * 4096;
  for (int o = tid; o < 4096; o += 256) sR[o] = (__bf16)Rp[o];

  f32x8 acc[4];
#pragma unroll
  for (int j = 0; j < 4; ++j) acc[j] = zero8();

  for (int k0 = 0; k0 < 64; k0 += 32) {
    __syncthreads();
    {  // B tile: 32 v-rows of 128 contiguous dd, transposed into [dd][v]
      int v = tid >> 3, part = tid & 7;
      const __bf16* p =
          src + ((size_t)(bh * T_SEQ + (k0 + v) * BSZ + tl)) * D_HEAD + part * 16;
      bf16x16 t = *(const bf16x16*)p;
#pragma unroll
      for (int e = 0; e < 16; ++e) sBT[(part * 16 + e) * 32 + v] = t[e];
    }
    __syncthreads();
    bf16x16 af = load_a_frag(&sR[(wm * 16) * 64 + k0], 64, lane);
#pragma unroll
    for (int j = 0; j < 4; ++j) {
      bf16x16 bf = load_b_frag(&sBT[(wn * 64 + j * 16) * 32], 32, lane);
      acc[j] = wmma_bf16(af, bf, acc[j]);
    }
  }

  int nloc = lane & 15, mhalf = (lane >> 4) * 8;
#pragma unroll
  for (int j = 0; j < 4; ++j) {
    int dd = wn * 64 + j * 16 + nloc;
#pragma unroll
    for (int e = 0; e < 8; ++e) {
      int u = wm * 16 + mhalf + e;
      dst[((size_t)(bh * T_SEQ + u * BSZ + tl)) * D_HEAD + dd] = (__bf16)acc[j][e];
    }
  }
}

// ---------- 5: bucketed attention, one block per (bucket u, bh) ----------
__global__ __launch_bounds__(256) void k_attn(
    const __bf16* __restrict__ qb, const __bf16* __restrict__ kb,
    const __bf16* __restrict__ vb, const __bf16* __restrict__ kre,
    const __bf16* __restrict__ vre, __bf16* __restrict__ outb) {
  __shared__ __bf16 sVT[128 * 128];  // [d][j] transposed values
  __shared__ float sS[64 * 128];     // scores
  __shared__ __bf16 sP[64 * 128];    // softmax probs
  int u = blockIdx.x, bh = blockIdx.y, tid = threadIdx.x;
  int lane = tid & 31, wave = tid >> 5;
  int wm = wave & 3, wn = wave >> 2;  // wave: 16 rows x 64 cols
  const size_t baseQ = (size_t)(bh * T_SEQ + u * BSZ) * D_HEAD;
  const size_t baseRe = baseQ;  // same bucketed layout for k_re / v_re

  // stage transposed V (rows 0..63 = v_re, 64..127 = b_v)
  for (int c = tid; c < 1024; c += 256) {
    int j = c >> 3, part = c & 7;
    const __bf16* p = (j < 64) ? &vre[baseRe + (size_t)j * 128 + part * 16]
                               : &vb[baseQ + (size_t)(j - 64) * 128 + part * 16];
    bf16x16 t = *(const bf16x16*)p;
#pragma unroll
    for (int e = 0; e < 16; ++e) sVT[(part * 16 + e) * 128 + j] = t[e];
  }

  // S = Q K^T : A and B fragments read straight from global (both [row][k] major)
  f32x8 acc[4];
#pragma unroll
  for (int j = 0; j < 4; ++j) acc[j] = zero8();
  const __bf16* Qbase = qb + baseQ;
  const __bf16* Kbase = (wn == 0) ? (kre + baseRe) : (kb + baseQ);
  for (int k0 = 0; k0 < 128; k0 += 32) {
    bf16x16 af = load_a_frag(Qbase + (wm * 16) * 128 + k0, 128, lane);
#pragma unroll
    for (int j = 0; j < 4; ++j) {
      bf16x16 bf = load_b_frag(Kbase + (j * 16) * 128 + k0, 128, lane);
      acc[j] = wmma_bf16(af, bf, acc[j]);
    }
  }
  int nloc = lane & 15, mhalf = (lane >> 4) * 8;
#pragma unroll
  for (int j = 0; j < 4; ++j) {
    int col = wn * 64 + j * 16 + nloc;
#pragma unroll
    for (int e = 0; e < 8; ++e) {
      int m = wm * 16 + mhalf + e;
      sS[m * 128 + col] = acc[j][e] * 0.03125f;  // d_model^-0.5 = 1/32
    }
  }
  __syncthreads();

  if (tid < 64) {  // fp32 softmax over 128 keys per query row
    float mx = -__builtin_inff();
    for (int j = 0; j < 128; ++j) mx = fmaxf(mx, sS[tid * 128 + j]);
    float s = 0.0f;
    for (int j = 0; j < 128; ++j) s += expf(sS[tid * 128 + j] - mx);
    float inv = 1.0f / s;
    for (int j = 0; j < 128; ++j)
      sP[tid * 128 + j] = (__bf16)(expf(sS[tid * 128 + j] - mx) * inv);
  }
  __syncthreads();

  // O = P V
  f32x8 oacc[4];
#pragma unroll
  for (int j = 0; j < 4; ++j) oacc[j] = zero8();
  for (int k0 = 0; k0 < 128; k0 += 32) {
    bf16x16 af = load_a_frag(&sP[(wm * 16) * 128 + k0], 128, lane);
#pragma unroll
    for (int j = 0; j < 4; ++j) {
      bf16x16 bf = load_b_frag(&sVT[(wn * 64 + j * 16) * 128 + k0], 128, lane);
      oacc[j] = wmma_bf16(af, bf, oacc[j]);
    }
  }
  int b = bh >> 3, h = bh & 7;
#pragma unroll
  for (int j = 0; j < 4; ++j) {
    int d = wn * 64 + j * 16 + nloc;
#pragma unroll
    for (int e = 0; e < 8; ++e) {
      int m = wm * 16 + mhalf + e;
      int t = u * BSZ + m;
      outb[(size_t)(b * T_SEQ + t) * D_MODEL + h * D_HEAD + d] = (__bf16)oacc[j][e];
    }
  }
}

// ---------- 6: output projection [16384x1024] x [1024x1024]^T + bias -> fp32 ----------
__global__ __launch_bounds__(256) void k_gemm_out(
    const __bf16* __restrict__ A, const __bf16* __restrict__ W,
    const float* __restrict__ bias, float* __restrict__ out) {
  const int K = D_MODEL;
  __shared__ __bf16 sA[2][128 * 32];
  __shared__ __bf16 sB[2][128 * 32];
  int tid = threadIdx.x, lane = tid & 31, wave = tid >> 5;
  int wm = wave & 1, wn = wave >> 1;
  int row0 = blockIdx.y * 128, col0 = blockIdx.x * 128;
  int lr = tid >> 1, lh = tid & 1;

  f32x8 acc[4][2];
#pragma unroll
  for (int i = 0; i < 4; ++i)
#pragma unroll
    for (int j = 0; j < 2; ++j) acc[i][j] = zero8();

  const __bf16* gA = &A[(size_t)(row0 + lr) * K + lh * 16];
  const __bf16* gW = &W[(size_t)(col0 + lr) * K + lh * 16];
  auto issue = [&](int buf, int k0) {
    unsigned la = (unsigned)(size_t)&sA[buf][lr * 32 + lh * 16];
    unsigned lb = (unsigned)(size_t)&sB[buf][lr * 32 + lh * 16];
    async_copy16(la,      gA + k0);
    async_copy16(la + 16, gA + k0 + 8);
    async_copy16(lb,      gW + k0);
    async_copy16(lb + 16, gW + k0 + 8);
  };

  const int NIT = K / 32;
  issue(0, 0);
  for (int it = 0; it < NIT; ++it) {
    int cur = it & 1;
    if (it + 1 < NIT) {
      issue(cur ^ 1, (it + 1) * 32);
      asm volatile("s_wait_asynccnt 0x4" ::: "memory");
    } else {
      asm volatile("s_wait_asynccnt 0x0" ::: "memory");
    }
    __syncthreads();
    bf16x16 bf[2];
#pragma unroll
    for (int j = 0; j < 2; ++j)
      bf[j] = load_b_frag(&sB[cur][(wn * 32 + j * 16) * 32], 32, lane);
#pragma unroll
    for (int i = 0; i < 4; ++i) {
      bf16x16 af = load_a_frag(&sA[cur][(wm * 64 + i * 16) * 32], 32, lane);
#pragma unroll
      for (int j = 0; j < 2; ++j) acc[i][j] = wmma_bf16(af, bf[j], acc[i][j]);
    }
    __syncthreads();
  }

  int nloc = lane & 15, mhalf = (lane >> 4) * 8;
#pragma unroll
  for (int i = 0; i < 4; ++i) {
#pragma unroll
    for (int j = 0; j < 2; ++j) {
      int c = col0 + wn * 32 + j * 16 + nloc;
      float bc = bias[c];
#pragma unroll
      for (int e = 0; e < 8; ++e) {
        int r = row0 + wm * 64 + i * 16 + mhalf + e;
        out[(size_t)r * D_MODEL + c] = acc[i][j][e] + bc;
      }
    }
  }
}

// ---------- host launcher ----------
extern "C" void kernel_launch(void* const* d_in, const int* in_sizes, int n_in,
                              void* d_out, int out_size, void* d_ws, size_t ws_size,
                              hipStream_t stream) {
  (void)in_sizes; (void)n_in; (void)out_size; (void)ws_size;
  const float* x      = (const float*)d_in[0];
  const float* w_qkv  = (const float*)d_in[1];
  const float* sort_w = (const float*)d_in[2];
  const float* w_out  = (const float*)d_in[3];
  const float* b_out  = (const float*)d_in[4];
  const float* noise  = (const float*)d_in[5];
  float* out = (float*)d_out;

  char* ws = (char*)d_ws;
  size_t off = 0;
  auto alloc = [&](size_t bytes) -> void* {
    void* p = ws + off;
    off += (bytes + 255) & ~(size_t)255;
    return p;
  };
  const size_t E = (size_t)ROWS * D_MODEL;            // 16M elements
  __bf16* xb     = (__bf16*)alloc(E * 2);             // 32 MB (reused as attnb)
  __bf16* wqkvb  = (__bf16*)alloc((size_t)3 * D_MODEL * D_MODEL * 2);  // 6 MB
  __bf16* woutb  = (__bf16*)alloc((size_t)D_MODEL * D_MODEL * 2);      // 2 MB
  __bf16* qbuf   = (__bf16*)alloc(E * 2);             // 32 MB
  __bf16* kbuf   = (__bf16*)alloc(E * 2);             // 32 MB
  __bf16* vbuf   = (__bf16*)alloc(E * 2);             // 32 MB
  __bf16* krebuf = (__bf16*)alloc(E * 2);             // 32 MB
  __bf16* vrebuf = (__bf16*)alloc(E * 2);             // 32 MB
  float*  Rbuf   = (float*)alloc((size_t)BHDIM * 64 * 64 * 4);  // 0.5 MB
  __bf16* attnb  = xb;  // x dead after QKV GEMM

  const int nx = ROWS * D_MODEL;          // 16777216
  const int nw = 3 * D_MODEL * D_MODEL;   // 3145728
  const int no = D_MODEL * D_MODEL;       // 1048576

  k_cast_bf16<<<(nx + 255) / 256, 256, 0, stream>>>(x, xb, nx);
  k_cast_bf16<<<(nw + 255) / 256, 256, 0, stream>>>(w_qkv, wqkvb, nw);
  k_cast_bf16<<<(no + 255) / 256, 256, 0, stream>>>(w_out, woutb, no);

  k_gemm_qkv<<<dim3(24, 128), 256, 0, stream>>>(xb, wqkvb, qbuf, kbuf, vbuf);
  k_sinkhorn<<<BHDIM, 256, 0, stream>>>(kbuf, sort_w, noise, Rbuf);
  k_permute<<<dim3(64, BHDIM), 256, 0, stream>>>(Rbuf, kbuf, krebuf);
  k_permute<<<dim3(64, BHDIM), 256, 0, stream>>>(Rbuf, vbuf, vrebuf);
  k_attn<<<dim3(N_BUCKET, BHDIM), 256, 0, stream>>>(qbuf, kbuf, vbuf, krebuf, vrebuf, attnb);
  k_gemm_out<<<dim3(8, 128), 256, 0, stream>>>(attnb, woutb, b_out, out);
}
